// CoAttention_65395172049135
// MI455X (gfx1250) — matrix operand order repaired and checked by reading, not compiled, not run on hardware
//
#include <hip/hip_runtime.h>

// ---------------- CDNA5 WMMA bf16 infrastructure ----------------
typedef __attribute__((ext_vector_type(16))) __bf16 v16bf;
typedef __attribute__((ext_vector_type(8)))  float  v8f;

#define B_N   64
#define D_N   512
#define CL_N  1024
#define QL_N  128
#define NEGC  (-1.0e15f)

static __device__ __forceinline__ unsigned short f2bf(float x) {
    unsigned u = __float_as_uint(x);
    u += 0x7FFFu + ((u >> 16) & 1u);           // round-to-nearest-even
    return (unsigned short)(u >> 16);
}
static __device__ __forceinline__ float bf2f(unsigned short h) {
    return __uint_as_float(((unsigned)h) << 16);
}

union Frag { v16bf v; uint4 u[2]; };

// A fragment: row-major MxK, per-lane K = {hl*8..hl*8+7, 16+hl*8..16+hl*8+7}
static __device__ __forceinline__ v16bf ld_a(const unsigned short* p, int hl) {
    Frag f;
    f.u[0] = *(const uint4*)(p + hl * 8);
    f.u[1] = *(const uint4*)(p + 16 + hl * 8);
    return f.v;
}
// B fragment: B^T row-major NxK, per-lane K = {hl*8..hl*8+7, 16+hl*8..16+hl*8+7}
static __device__ __forceinline__ v16bf ld_b(const unsigned short* p, int hl) {
    Frag f;
    f.u[0] = *(const uint4*)(p + hl * 8);
    f.u[1] = *(const uint4*)(p + 16 + hl * 8);
    return f.v;
}
#define WMMA_BF16(a, b, c) \
    __builtin_amdgcn_wmma_f32_16x16x32_bf16(false, (a), false, (b), (short)0, (c), false, false)

static __device__ __forceinline__ float wave_max(float v) {
    #pragma unroll
    for (int o = 16; o > 0; o >>= 1) v = fmaxf(v, __shfl_xor(v, o, 32));
    return v;
}
static __device__ __forceinline__ float wave_sum(float v) {
    #pragma unroll
    for (int o = 16; o > 0; o >>= 1) v += __shfl_xor(v, o, 32);
    return v;
}

// ---------------- Kernel 1: C -> Cob (bf16, [b,d,i]) + Ctb (bf16, [b,i,d]) ----------------
__global__ void k_conv_transpose_c(const float* __restrict__ C,
                                   unsigned short* __restrict__ Cob,
                                   unsigned short* __restrict__ Ctb) {
    __shared__ float tile[32][33];
    const int b = blockIdx.z, i0 = blockIdx.x * 32, d0 = blockIdx.y * 32;
    const int tx = threadIdx.x, ty = threadIdx.y;
    const size_t bo = (size_t)b * D_N * CL_N;
    #pragma unroll
    for (int r = 0; r < 4; ++r) {
        int dd = ty + r * 8;
        float v = C[bo + (size_t)(d0 + dd) * CL_N + i0 + tx];
        Cob[bo + (size_t)(d0 + dd) * CL_N + i0 + tx] = f2bf(v);
        tile[dd][tx] = v;
    }
    __syncthreads();
    const size_t bt = (size_t)b * CL_N * D_N;
    #pragma unroll
    for (int r = 0; r < 4; ++r) {
        int ii = ty + r * 8;
        Ctb[bt + (size_t)(i0 + ii) * D_N + d0 + tx] = f2bf(tile[tx][ii]);
    }
}

// ---------------- Kernel 2: Q -> Qob (bf16 [b,d,j]) + Qmb (w_m*Q_t, bf16 [b,j,d]) -------
__global__ void k_conv_transpose_q(const float* __restrict__ Q,
                                   const float* __restrict__ W0,
                                   unsigned short* __restrict__ Qob,
                                   unsigned short* __restrict__ Qmb) {
    __shared__ float tile[32][33];
    const int b = blockIdx.z, j0 = blockIdx.x * 32, d0 = blockIdx.y * 32;
    const int tx = threadIdx.x, ty = threadIdx.y;
    const size_t bo = (size_t)b * D_N * QL_N;
    #pragma unroll
    for (int r = 0; r < 4; ++r) {
        int dd = ty + r * 8;
        float v = Q[bo + (size_t)(d0 + dd) * QL_N + j0 + tx];
        Qob[bo + (size_t)(d0 + dd) * QL_N + j0 + tx] = f2bf(v);
        tile[dd][tx] = v;
    }
    __syncthreads();
    const float wm = W0[2 * D_N + d0 + tx];
    const size_t bt = (size_t)b * QL_N * D_N;
    #pragma unroll
    for (int r = 0; r < 4; ++r) {
        int jj = ty + r * 8;
        Qmb[bt + (size_t)(j0 + jj) * D_N + d0 + tx] = f2bf(tile[tx][jj] * wm);
    }
}

// ---------------- Kernel 3: Wr f32 -> bf16 ----------------
__global__ void k_conv_wr(const float* __restrict__ Wr, unsigned short* __restrict__ Wrb, int n) {
    int i = blockIdx.x * blockDim.x + threadIdx.x;
    if (i < n) Wrb[i] = f2bf(Wr[i]);
}

// ---------------- Kernel 4: cdot[b,i] = C_t[b,i,:] . w_c ----------------
__global__ void k_cdot(const unsigned short* __restrict__ Ctb,
                       const float* __restrict__ W0, float* __restrict__ cdot) {
    int wave = blockIdx.x * 8 + threadIdx.x / 32;
    int lane = threadIdx.x & 31;
    int b = wave / CL_N, i = wave % CL_N;
    const unsigned short* row = Ctb + (size_t)b * CL_N * D_N + (size_t)i * D_N;
    float s = 0.f;
    #pragma unroll
    for (int t = 0; t < 16; ++t) {
        int d = lane + 32 * t;
        s += bf2f(row[d]) * W0[d];
    }
    s = wave_sum(s);
    if (lane == 0) cdot[b * CL_N + i] = s;
}

// ---------------- Kernel 5: qdot[b,j] = Q_t[b,j,:] . w_q (from f32 Q) ----------------
__global__ void k_qdot(const float* __restrict__ Q,
                       const float* __restrict__ W0, float* __restrict__ qdot) {
    int wave = blockIdx.x * 8 + threadIdx.x / 32;
    int lane = threadIdx.x & 31;
    int b = wave / QL_N, j = wave % QL_N;
    const float* base = Q + (size_t)b * D_N * QL_N;
    float s = 0.f;
    #pragma unroll
    for (int t = 0; t < 16; ++t) {
        int d = lane + 32 * t;
        s += base[(size_t)d * QL_N + j] * W0[D_N + d];
    }
    s = wave_sum(s);
    if (lane == 0) qdot[b * QL_N + j] = s;
}

// ------- Kernel 6: S = Ctb @ Qmb^T + dots; row-softmax -> Sbar bf16; raw S -> f32 -------
__global__ void k_s_softmax(const unsigned short* __restrict__ Ctb,
                            const unsigned short* __restrict__ Qmb,
                            const float* __restrict__ cdot,
                            const float* __restrict__ qdot,
                            const float* __restrict__ q_mask,
                            float* __restrict__ Sbuf,
                            unsigned short* __restrict__ Sbar) {
    __shared__ float Sl[64][132];
    __shared__ float qd_s[QL_N], qm_s[QL_N];
    const int b = blockIdx.y, i0 = blockIdx.x * 64;
    const int tid = threadIdx.x, w = tid / 32, lane = tid & 31;
    const int hl = lane >> 4, r16 = lane & 15;
    const int mstrip = w >> 1, nhalf = w & 1;

    if (tid < QL_N) { qd_s[tid] = qdot[b * QL_N + tid]; qm_s[tid] = q_mask[b * QL_N + tid]; }

    const unsigned short* Arow = Ctb + (size_t)b * CL_N * D_N + (size_t)(i0 + mstrip * 16 + r16) * D_N;
    const unsigned short* Bb   = Qmb + (size_t)b * QL_N * D_N;

    v8f acc[4] = {};
    for (int k = 0; k < D_N; k += 32) {
        v16bf af = ld_a(Arow + k, hl);
        #pragma unroll
        for (int t = 0; t < 4; ++t) {
            int nr = nhalf * 64 + t * 16 + r16;
            v16bf bf = ld_b(Bb + (size_t)nr * D_N + k, hl);
            acc[t] = WMMA_BF16(af, bf, acc[t]);
        }
    }
    #pragma unroll
    for (int t = 0; t < 4; ++t)
        #pragma unroll
        for (int v = 0; v < 8; ++v)
            Sl[mstrip * 16 + v + 8 * hl][nhalf * 64 + t * 16 + r16] = acc[t][v];
    __syncthreads();

    if (tid < 64) {
        const int i = i0 + tid;
        const float cd = cdot[b * CL_N + i];
        float* Srow = Sbuf + (size_t)b * CL_N * QL_N + (size_t)i * QL_N;
        float mx = -3.0e38f;
        for (int j = 0; j < QL_N; ++j) {
            float s = Sl[tid][j] + cd + qd_s[j];
            Srow[j] = s;                                     // raw S for column softmax
            float qm = qm_s[j];
            float msk = s * qm + NEGC * (1.0f - qm);
            Sl[tid][j] = msk;
            mx = fmaxf(mx, msk);
        }
        float sum = 0.f;
        for (int j = 0; j < QL_N; ++j) { float e = __expf(Sl[tid][j] - mx); Sl[tid][j] = e; sum += e; }
        float inv = 1.0f / sum;
        unsigned short* Srb = Sbar + (size_t)b * CL_N * QL_N + (size_t)i * QL_N;
        for (int j = 0; j < QL_N; ++j) Srb[j] = f2bf(Sl[tid][j] * inv);
    }
}

// ---------------- Kernel 7: column softmax -> SbbT bf16 [b,j,i] ----------------
__global__ void k_colsoftmax(const float* __restrict__ Sbuf,
                             const float* __restrict__ c_mask,
                             unsigned short* __restrict__ SbbT) {
    int wave = blockIdx.x * 8 + threadIdx.x / 32;
    int lane = threadIdx.x & 31;
    int b = wave / QL_N, j = wave % QL_N;
    const float* Sb = Sbuf + (size_t)b * CL_N * QL_N;
    const float* cm = c_mask + b * CL_N;
    float loc[32];
    float mx = -3.0e38f;
    #pragma unroll
    for (int t = 0; t < 32; ++t) {
        int i = lane + 32 * t;
        float s = Sb[(size_t)i * QL_N + j];
        float m = cm[i];
        float v = s * m + NEGC * (1.0f - m);
        loc[t] = v;
        mx = fmaxf(mx, v);
    }
    mx = wave_max(mx);
    float sum = 0.f;
    #pragma unroll
    for (int t = 0; t < 32; ++t) { loc[t] = __expf(loc[t] - mx); sum += loc[t]; }
    sum = wave_sum(sum);
    float inv = 1.0f / sum;
    unsigned short* out = SbbT + (size_t)b * QL_N * CL_N + (size_t)j * CL_N;
    #pragma unroll
    for (int t = 0; t < 32; ++t) out[lane + 32 * t] = f2bf(loc[t] * inv);
}

// ---------------- Kernel 8: Tt[b,d,j] = (SbbT @ C_t) transposed-store ----------------
__global__ void k_t_gemm(const unsigned short* __restrict__ SbbT,
                         const unsigned short* __restrict__ Cob,
                         unsigned short* __restrict__ Tt) {
    const int b = blockIdx.y, n0 = blockIdx.x * 64;
    const int tid = threadIdx.x, w = tid / 32, lane = tid & 31;
    const int hl = lane >> 4, r16 = lane & 15;
    const unsigned short* Arow = SbbT + (size_t)b * QL_N * CL_N + (size_t)(w * 16 + r16) * CL_N;
    const unsigned short* Bb   = Cob  + (size_t)b * D_N * CL_N;
    v8f acc[4] = {};
    for (int k = 0; k < CL_N; k += 32) {
        v16bf af = ld_a(Arow + k, hl);
        #pragma unroll
        for (int t = 0; t < 4; ++t) {
            int nr = n0 + t * 16 + r16;
            v16bf bf = ld_b(Bb + (size_t)nr * CL_N + k, hl);
            acc[t] = WMMA_BF16(af, bf, acc[t]);
        }
    }
    #pragma unroll
    for (int t = 0; t < 4; ++t) {
        int dIdx = n0 + t * 16 + r16;
        union { uint4 u; unsigned short s[8]; } pk;
        #pragma unroll
        for (int v = 0; v < 8; ++v) pk.s[v] = f2bf(acc[t][v]);
        *(uint4*)(Tt + (size_t)b * D_N * QL_N + (size_t)dIdx * QL_N + w * 16 + 8 * hl) = pk.u;
    }
}

// ------- Kernel 9: A = Sbar@Q_t, Bm = Sbar@T ; write Ab, C*A, C*Bm (bf16) -------
__global__ void k_abm_gemm(const unsigned short* __restrict__ Sbar,
                           const unsigned short* __restrict__ Qob,
                           const unsigned short* __restrict__ Tt,
                           const unsigned short* __restrict__ Ctb,
                           unsigned short* __restrict__ Ab,
                           unsigned short* __restrict__ CAb,
                           unsigned short* __restrict__ CBb) {
    const int b = blockIdx.z, n0 = blockIdx.y * 64;
    const int tid = threadIdx.x, w = tid / 32, lane = tid & 31;
    const int hl = lane >> 4, r16 = lane & 15;
    const int iBase = blockIdx.x * 128 + w * 16;
    const unsigned short* Arow = Sbar + (size_t)b * CL_N * QL_N + (size_t)(iBase + r16) * QL_N;
    const unsigned short* Bq   = Qob + (size_t)b * D_N * QL_N;
    const unsigned short* Bt   = Tt  + (size_t)b * D_N * QL_N;
    v8f accA[4] = {}, accB[4] = {};
    for (int k = 0; k < QL_N; k += 32) {
        v16bf af = ld_a(Arow + k, hl);
        #pragma unroll
        for (int t = 0; t < 4; ++t) {
            int nr = n0 + t * 16 + r16;
            v16bf bq = ld_b(Bq + (size_t)nr * QL_N + k, hl);
            accA[t] = WMMA_BF16(af, bq, accA[t]);
            v16bf bt = ld_b(Bt + (size_t)nr * QL_N + k, hl);
            accB[t] = WMMA_BF16(af, bt, accB[t]);
        }
    }
    const size_t bo = (size_t)b * CL_N * D_N;
    #pragma unroll
    for (int t = 0; t < 4; ++t) {
        int dcol = n0 + t * 16 + r16;
        #pragma unroll
        for (int v = 0; v < 8; ++v) {
            int i = iBase + v + 8 * hl;
            size_t idx = bo + (size_t)i * D_N + dcol;
            float ct = bf2f(Ctb[idx]);
            float a = accA[t][v], bm = accB[t][v];
            Ab[idx]  = f2bf(a);
            CAb[idx] = f2bf(ct * a);
            CBb[idx] = f2bf(ct * bm);
        }
    }
}

// ------- Kernel 10: Out[b,o,c] = Wr(512x2048) @ [C_t | A | C*A | C*Bm]^T -------
__global__ void k_final_gemm(const unsigned short* __restrict__ Wrb,
                             const unsigned short* __restrict__ Ctb,
                             const unsigned short* __restrict__ Ab,
                             const unsigned short* __restrict__ CAb,
                             const unsigned short* __restrict__ CBb,
                             float* __restrict__ Out) {
    const int b = blockIdx.z, c0 = blockIdx.y * 64;
    const int tid = threadIdx.x, w = tid / 32, lane = tid & 31;
    const int hl = lane >> 4, r16 = lane & 15;
    const int oBase = blockIdx.x * 128 + w * 16;
    const unsigned short* Arow = Wrb + (size_t)(oBase + r16) * (4 * D_N);
    const size_t bo = (size_t)b * CL_N * D_N;
    const unsigned short* segs[4] = { Ctb + bo, Ab + bo, CAb + bo, CBb + bo };
    v8f acc[4] = {};
    for (int k = 0; k < 4 * D_N; k += 32) {
        const unsigned short* segBase = segs[k >> 9];
        const int kk = k & (D_N - 1);
        v16bf af = ld_a(Arow + k, hl);
        #pragma unroll
        for (int t = 0; t < 4; ++t) {
            int cr = c0 + t * 16 + r16;
            v16bf bf = ld_b(segBase + (size_t)cr * D_N + kk, hl);
            acc[t] = WMMA_BF16(af, bf, acc[t]);
        }
    }
    float* Ob = Out + (size_t)b * D_N * CL_N;
    #pragma unroll
    for (int t = 0; t < 4; ++t) {
        int c = c0 + t * 16 + r16;
        #pragma unroll
        for (int v = 0; v < 8; ++v) {
            int o = oBase + v + 8 * hl;
            Ob[(size_t)o * CL_N + c] = acc[t][v];
        }
    }
}

// ---------------- Host launcher ----------------
extern "C" void kernel_launch(void* const* d_in, const int* in_sizes, int n_in,
                              void* d_out, int out_size, void* d_ws, size_t ws_size,
                              hipStream_t stream) {
    const float* C      = (const float*)d_in[0];
    const float* Q      = (const float*)d_in[1];
    const float* c_mask = (const float*)d_in[2];
    const float* q_mask = (const float*)d_in[3];
    const float* W0     = (const float*)d_in[4];
    const float* Wr     = (const float*)d_in[5];
    float* Out = (float*)d_out;

    char* ws = (char*)d_ws;
    size_t off = 0;
    auto alloc = [&](size_t bytes) { char* p = ws + off; off = (off + bytes + 255) & ~(size_t)255; return p; };

    unsigned short* Ctb  = (unsigned short*)alloc((size_t)B_N * CL_N * D_N * 2);  // C_t bf16 [b,i,d]
    unsigned short* Cob  = (unsigned short*)alloc((size_t)B_N * D_N * CL_N * 2);  // C   bf16 [b,d,i]
    unsigned short* Qmb  = (unsigned short*)alloc((size_t)B_N * QL_N * D_N * 2);  // w_m*Q_t  [b,j,d]
    unsigned short* Qob  = (unsigned short*)alloc((size_t)B_N * D_N * QL_N * 2);  // Q   bf16 [b,d,j]
    float*          Sbuf = (float*)        alloc((size_t)B_N * CL_N * QL_N * 4);  // raw S
    unsigned short* Sbar = (unsigned short*)alloc((size_t)B_N * CL_N * QL_N * 2);
    unsigned short* SbbT = (unsigned short*)alloc((size_t)B_N * QL_N * CL_N * 2);
    unsigned short* Tt   = (unsigned short*)alloc((size_t)B_N * D_N * QL_N * 2);
    unsigned short* Ab   = (unsigned short*)alloc((size_t)B_N * CL_N * D_N * 2);
    unsigned short* CAb  = (unsigned short*)alloc((size_t)B_N * CL_N * D_N * 2);
    unsigned short* CBb  = (unsigned short*)alloc((size_t)B_N * CL_N * D_N * 2);
    float*          cdot = (float*)alloc((size_t)B_N * CL_N * 4);
    float*          qdot = (float*)alloc((size_t)B_N * QL_N * 4);
    unsigned short* Wrb  = (unsigned short*)alloc((size_t)D_N * 4 * D_N * 2);

    k_conv_transpose_c<<<dim3(CL_N / 32, D_N / 32, B_N), dim3(32, 8), 0, stream>>>(C, Cob, Ctb);
    k_conv_transpose_q<<<dim3(QL_N / 32, D_N / 32, B_N), dim3(32, 8), 0, stream>>>(Q, W0, Qob, Qmb);
    k_conv_wr<<<(D_N * 4 * D_N + 255) / 256, 256, 0, stream>>>(Wr, Wrb, D_N * 4 * D_N);
    k_cdot<<<(B_N * CL_N) / 8, 256, 0, stream>>>(Ctb, W0, cdot);
    k_qdot<<<(B_N * QL_N) / 8, 256, 0, stream>>>(Q, W0, qdot);
    k_s_softmax<<<dim3(CL_N / 64, B_N), 256, 0, stream>>>(Ctb, Qmb, cdot, qdot, q_mask, Sbuf, Sbar);
    k_colsoftmax<<<(B_N * QL_N) / 8, 256, 0, stream>>>(Sbuf, c_mask, SbbT);
    k_t_gemm<<<dim3(D_N / 64, B_N), 256, 0, stream>>>(SbbT, Cob, Tt);
    k_abm_gemm<<<dim3(CL_N / 128, D_N / 64, B_N), 256, 0, stream>>>(Sbar, Qob, Tt, Ctb, Ab, CAb, CBb);
    k_final_gemm<<<dim3(D_N / 128, CL_N / 64, B_N), 256, 0, stream>>>(Wrb, Ctb, Ab, CAb, CBb, Out);
}
